// GAT_68289980006593
// MI455X (gfx1250) — matrix operand order repaired and checked by reading, not compiled, not run on hardware
//
#include <hip/hip_runtime.h>
#include <hip/hip_bf16.h>
#include <stdint.h>

typedef __attribute__((ext_vector_type(16))) __bf16 v16bf;
typedef __attribute__((ext_vector_type(8)))  __bf16 v8bf;
typedef __attribute__((ext_vector_type(8)))  float  v8f;

#define HEADS 4
#define HID 64
#define HH 256      // HEADS*HID
#define IND 128     // IN_DIM
#define OUTD 128    // OUT_DIM
#define NEG_SLOPE 0.2f

// ---------------- small utility kernels ----------------

__global__ void k_f32_to_bf16(const float* __restrict__ in, __bf16* __restrict__ out, int n) {
  int i = blockIdx.x * blockDim.x + threadIdx.x;
  if (i < n) out[i] = (__bf16)in[i];
}

// in: [R,C] f32 row-major  ->  out: [C,R] bf16 row-major (columns contiguous)
__global__ void k_transpose_bf16(const float* __restrict__ in, __bf16* __restrict__ out,
                                 int R, int C) {
  int i = blockIdx.x * blockDim.x + threadIdx.x;
  if (i >= R * C) return;
  int r = i / C, c = i % C;
  out[(size_t)c * R + r] = (__bf16)in[(size_t)r * C + c];
}

__global__ void k_zero_f32(float* __restrict__ p, int n) {
  int i = blockIdx.x * blockDim.x + threadIdx.x;
  if (i < n) p[i] = 0.0f;
}

// ---------------- WMMA bf16 GEMM: C[M,Ncols] = A[M,K] * Bt[Ncols,K]^T (+bias) ----------------
// One wave per 16x16 output tile. Per-lane fragment layout follows the CDNA5
// 16-bit A (16x32) and B (32x16) VGPR striping (CDNA5 ISA 7.12.2).
__global__ void __launch_bounds__(128)
k_gemm_bf16_wmma(const __bf16* __restrict__ A, const __bf16* __restrict__ Bt,
                 float* __restrict__ C, const float* __restrict__ bias,
                 int M, int Ncols, int K) {
  const int wid    = blockIdx.x * (blockDim.x >> 5) + (threadIdx.x >> 5);
  const int ntiles = Ncols >> 4;
  const int tileM  = wid / ntiles;
  const int tileN  = wid - tileM * ntiles;
  if (tileM * 16 >= M) return;                 // uniform per wave -> EXEC stays all-1s
  const int lane = threadIdx.x & 31;
  const int half = lane >> 4;                  // lane group 0-15 vs 16-31
  const int m    = lane & 15;
  const size_t arow = (size_t)(tileM * 16 + m) * (size_t)K;
  const size_t brow = (size_t)(tileN * 16 + m) * (size_t)K;

  v8f acc = {};
  for (int k0 = 0; k0 < K; k0 += 32) {
    // A fragment: lane half 0 holds K {k0..k0+7, k0+16..k0+23}; half 1 -> +8
    v8bf alo = *(const v8bf*)(A + arow + k0 + half * 8);
    v8bf ahi = *(const v8bf*)(A + arow + k0 + half * 8 + 16);
    // B fragment: lane half 0 holds K {k0..k0+7, k0+16..k0+23}; half 1 -> +8
    v8bf blo = *(const v8bf*)(Bt + brow + k0 + half * 8);
    v8bf bhi = *(const v8bf*)(Bt + brow + k0 + half * 8 + 16);
    v16bf a, b;
#pragma unroll
    for (int e = 0; e < 8; ++e) {
      a[e] = alo[e]; a[e + 8] = ahi[e];
      b[e] = blo[e]; b[e + 8] = bhi[e];
    }
    acc = __builtin_amdgcn_wmma_f32_16x16x32_bf16(
        /*neg_a=*/false, a, /*neg_b=*/false, b,
        /*c_mod=*/(short)0, acc, /*reuse_a=*/false, /*reuse_b=*/false);
  }

  const int col = tileN * 16 + m;
  const float bv = bias ? bias[col] : 0.0f;
#pragma unroll
  for (int v = 0; v < 8; ++v) {                // C/D: VGPR v -> row v (+8 for upper lanes)
    int row = tileM * 16 + half * 8 + v;
    C[(size_t)row * Ncols + col] = acc[v] + bv;
  }
}

// ---------------- attention scores + softmax state init ----------------
__global__ void k_attn_scores(const float* __restrict__ h,
                              const float* __restrict__ att_src,
                              const float* __restrict__ att_dst,
                              float* __restrict__ as, float* __restrict__ ad,
                              float* __restrict__ mb, float* __restrict__ dn, int N) {
  int i = blockIdx.x * blockDim.x + threadIdx.x;
  if (i >= N * HEADS) return;
  int n = i >> 2, hd = i & 3;
  const float* hp = h + (size_t)n * HH + hd * HID;
  const float* ws = att_src + hd * HID;
  const float* wd = att_dst + hd * HID;
  float ss = 0.f, sd = 0.f;
#pragma unroll 8
  for (int f = 0; f < HID; ++f) { float v = hp[f]; ss += v * ws[f]; sd += v * wd[f]; }
  as[i] = ss; ad[i] = sd;
  mb[i] = -__builtin_inff();
  dn[i] = 0.0f;
}

// ---------------- edge passes ----------------
__device__ __forceinline__ void atomicMaxF(float* addr, float value) {
  if (value >= 0.0f) atomicMax((int*)addr, __float_as_int(value));
  else               atomicMin((unsigned int*)addr, __float_as_uint(value));
}

// pass A: e = leaky_relu(a_src[s]+a_dst[d]); store e; m[d] = max(m[d], e)
__global__ void k_edge_logits(const long long* __restrict__ ei,
                              const float* __restrict__ as, const float* __restrict__ ad,
                              float* __restrict__ ev, float* __restrict__ mb,
                              int E, int N) {
  int i = blockIdx.x * blockDim.x + threadIdx.x;
  if (i >= E + N) return;
  int s, d;
  if (i < E) { s = (int)ei[i]; d = (int)ei[(size_t)E + i]; }
  else       { s = d = i - E; }                         // self loops
#pragma unroll
  for (int hd = 0; hd < HEADS; ++hd) {
    float e = as[s * HEADS + hd] + ad[d * HEADS + hd];
    e = (e > 0.0f) ? e : NEG_SLOPE * e;
    ev[(size_t)i * HEADS + hd] = e;
    atomicMaxF(&mb[d * HEADS + hd], e);
  }
}

// pass B: ex = exp(e - m[d]); store ex; denom[d] += ex
__global__ void k_edge_expsum(const long long* __restrict__ ei,
                              float* __restrict__ ev, const float* __restrict__ mb,
                              float* __restrict__ dn, int E, int N) {
  int i = blockIdx.x * blockDim.x + threadIdx.x;
  if (i >= E + N) return;
  int d = (i < E) ? (int)ei[(size_t)E + i] : (i - E);
#pragma unroll
  for (int hd = 0; hd < HEADS; ++hd) {
    float ex = __expf(ev[(size_t)i * HEADS + hd] - mb[d * HEADS + hd]);
    ev[(size_t)i * HEADS + hd] = ex;
    atomicAdd(&dn[d * HEADS + hd], ex);
  }
}

// pass C: agg[d] += h[s] * alpha ; one 256-thread block per edge, coalesced rows
__global__ void __launch_bounds__(256)
k_edge_scatter(const long long* __restrict__ ei,
               const float* __restrict__ ev, const float* __restrict__ dn,
               const float* __restrict__ h, float* __restrict__ agg,
               int E, int N) {
  int i = blockIdx.x;
  if (i >= E + N) return;
  int s, d;
  if (i < E) { s = (int)ei[i]; d = (int)ei[(size_t)E + i]; }
  else       { s = d = i - E; }
  int t  = threadIdx.x;          // feature index 0..255
  int hd = t >> 6;               // head
  float alpha = ev[(size_t)i * HEADS + hd] / dn[d * HEADS + hd];
  float val = h[(size_t)s * HH + t] * alpha;
  atomicAdd(&agg[(size_t)d * HH + t], val);
}

// bias + relu + convert to bf16 for the final GEMM
__global__ void k_bias_relu_bf16(const float* __restrict__ agg, const float* __restrict__ bias,
                                 __bf16* __restrict__ mid, int n) {
  int i = blockIdx.x * blockDim.x + threadIdx.x;
  if (i >= n) return;
  int c = i & (HH - 1);
  float v = agg[i] + bias[c];
  mid[i] = (__bf16)fmaxf(v, 0.0f);
}

// ---------------- host driver ----------------
extern "C" void kernel_launch(void* const* d_in, const int* in_sizes, int n_in,
                              void* d_out, int out_size, void* d_ws, size_t ws_size,
                              hipStream_t stream) {
  const float*     x       = (const float*)d_in[0];
  const long long* ei      = (const long long*)d_in[1];   // int64 [2,E]
  const float*     W       = (const float*)d_in[2];
  const float*     att_src = (const float*)d_in[3];
  const float*     att_dst = (const float*)d_in[4];
  const float*     bias    = (const float*)d_in[5];
  const float*     fc_w    = (const float*)d_in[6];
  const float*     fc_b    = (const float*)d_in[7];
  float*           out     = (float*)d_out;

  const int N  = in_sizes[0] / IND;
  const int E  = in_sizes[1] / 2;
  const int EN = E + N;

  // ---- carve workspace (all offsets 256B aligned) ----
  char* ws = (char*)d_ws;
  size_t off = 0;
  auto carve = [&](size_t bytes) -> char* {
    char* p = ws + off;
    off += (bytes + 255) & ~(size_t)255;
    return p;
  };
  float*  h   = (float*)carve((size_t)N * HH * 4);        // 51.2 MB
  float*  agg = (float*)carve((size_t)N * HH * 4);        // 51.2 MB
  float*  ev  = (float*)carve((size_t)EN * HEADS * 4);    // 13.6 MB
  __bf16* xb  = (__bf16*)carve((size_t)N * IND * 2);      // 12.8 MB
  float*  as  = (float*)carve((size_t)N * HEADS * 4);
  float*  ad  = (float*)carve((size_t)N * HEADS * 4);
  float*  mb  = (float*)carve((size_t)N * HEADS * 4);
  float*  dn  = (float*)carve((size_t)N * HEADS * 4);
  __bf16* Wt  = (__bf16*)carve((size_t)IND * HH * 2);
  __bf16* fwt = (__bf16*)carve((size_t)HH * OUTD * 2);
  __bf16* mid = (__bf16*)h;   // reuse h region: h dead after edge scatter

  const int B = 256;
  auto g1 = [&](int n) { return (n + B - 1) / B; };

  // precision prep: bf16 operands for the two dense GEMMs
  k_f32_to_bf16<<<g1(N * IND), B, 0, stream>>>(x, xb, N * IND);
  k_transpose_bf16<<<g1(IND * HH), B, 0, stream>>>(W, Wt, IND, HH);
  k_transpose_bf16<<<g1(HH * OUTD), B, 0, stream>>>(fc_w, fwt, HH, OUTD);

  // GEMM1: h = x @ W   [N,128]x[128,256]
  {
    int tilesM = (N + 15) / 16, ntN = HH / 16;
    int total = tilesM * ntN;
    k_gemm_bf16_wmma<<<(total + 3) / 4, 128, 0, stream>>>(xb, Wt, h, nullptr, N, HH, IND);
  }

  k_attn_scores<<<g1(N * HEADS), B, 0, stream>>>(h, att_src, att_dst, as, ad, mb, dn, N);
  k_zero_f32<<<g1(N * HH), B, 0, stream>>>(agg, N * HH);

  k_edge_logits<<<g1(EN), B, 0, stream>>>(ei, as, ad, ev, mb, E, N);
  k_edge_expsum<<<g1(EN), B, 0, stream>>>(ei, ev, mb, dn, E, N);
  k_edge_scatter<<<EN, 256, 0, stream>>>(ei, ev, dn, h, agg, E, N);

  k_bias_relu_bf16<<<g1(N * HH), B, 0, stream>>>(agg, bias, mid, N * HH);

  // GEMM2: out = relu(agg+bias) @ fc_w + fc_b   [N,256]x[256,128]
  {
    int tilesM = (N + 15) / 16, ntN = OUTD / 16;
    int total = tilesM * ntN;
    k_gemm_bf16_wmma<<<(total + 3) / 4, 128, 0, stream>>>(mid, fwt, out, fc_b, N, OUTD, HH);
  }
}
